// SynthesisBlock_20289425506660
// MI455X (gfx1250) — matrix operand order repaired and checked by reading, not compiled, not run on hardware
//
#include <hip/hip_runtime.h>
#include <hip/hip_bf16.h>

typedef __attribute__((ext_vector_type(16))) _Float16 v16h;
typedef __attribute__((ext_vector_type(8)))  _Float16 v8h;
typedef __attribute__((ext_vector_type(8)))  float    v8f;

#define HWPIX 65536
#define WID 256

__device__ __forceinline__ float leaky_(float v) { return v >= 0.f ? v : 0.2f * v; }

// ---------------------------------------------------------------------------
// Style MLP: s = leaky(layernorm(w@mw.T + mb)*g + b); sc = w@asw.T+asb; bi = w@abw.T+abb
// grid = (B), block = I threads
// ---------------------------------------------------------------------------
__global__ void style_kernel(const float* __restrict__ w,
                             const float* __restrict__ mw, const float* __restrict__ mb_,
                             const float* __restrict__ g,  const float* __restrict__ bt,
                             const float* __restrict__ asw, const float* __restrict__ asb,
                             const float* __restrict__ abw, const float* __restrict__ abb,
                             int I,
                             float* __restrict__ s_out, float* __restrict__ sc_out,
                             float* __restrict__ bi_out) {
  __shared__ float red[128];
  int b = blockIdx.x;
  int i = threadIdx.x;
  const float* wb = w + b * 512;
  float y = mb_[i], sc = asb[i], bi = abb[i];
  for (int k = 0; k < 512; ++k) {
    float wk = wb[k];
    y  += wk * mw [i * 512 + k];
    sc += wk * asw[i * 512 + k];
    bi += wk * abw[i * 512 + k];
  }
  sc_out[b * I + i] = sc;
  bi_out[b * I + i] = bi;
  red[i] = y; __syncthreads();
  for (int s = I >> 1; s > 0; s >>= 1) { if (i < s) red[i] += red[i + s]; __syncthreads(); }
  float mu = red[0] / (float)I; __syncthreads();
  float d = y - mu;
  red[i] = d * d; __syncthreads();
  for (int s = I >> 1; s > 0; s >>= 1) { if (i < s) red[i] += red[i + s]; __syncthreads(); }
  float var = red[0] / (float)I;
  float sn = d * rsqrtf(var + 1e-5f) * g[i] + bt[i];
  s_out[b * I + i] = sn >= 0.f ? sn : 0.2f * sn;
}

// ---------------------------------------------------------------------------
// Bilinear 2x upsample (half-pixel, clamp) NCHW f32 [8,128,128,128] -> NHWC f16
// [8,256,256,128]; fused per-(b,c) sum/sumsq stats via f32 atomics.
// grid = (y=256, b=8), block = 256 (c = tid&127, x parity = tid>>7)
// ---------------------------------------------------------------------------
__global__ void upsample_kernel(const float* __restrict__ x,
                                _Float16* __restrict__ out,
                                float* __restrict__ stats) {
  int y = blockIdx.x, b = blockIdx.y;
  int tid = threadIdx.x;
  int c = tid & 127, xh = tid >> 7;
  float sy = y * 0.5f - 0.25f;
  int y0 = (int)floorf(sy);
  float fy = sy - (float)y0;
  int y0c = y0 < 0 ? 0 : y0;
  int y1c = (y0 + 1) > 127 ? 127 : (y0 + 1);
  const float* plane = x + ((size_t)(b * 128 + c)) * 16384;
  float sum = 0.f, sq = 0.f;
  for (int xo = xh; xo < 256; xo += 2) {
    float sx = xo * 0.5f - 0.25f;
    int x0 = (int)floorf(sx);
    float fx = sx - (float)x0;
    int x0c = x0 < 0 ? 0 : x0;
    int x1c = (x0 + 1) > 127 ? 127 : (x0 + 1);
    float v00 = plane[y0c * 128 + x0c], v01 = plane[y0c * 128 + x1c];
    float v10 = plane[y1c * 128 + x0c], v11 = plane[y1c * 128 + x1c];
    float v = (1.f - fy) * ((1.f - fx) * v00 + fx * v01)
            +        fy  * ((1.f - fx) * v10 + fx * v11);
    out[((size_t)(b * 256 + y) * 256 + xo) * 128 + c] = (_Float16)v;
    sum += v; sq += v * v;
  }
  atomicAdd(&stats[(b * 128 + c) * 2],     sum);
  atomicAdd(&stats[(b * 128 + c) * 2 + 1], sq);
}

// ---------------------------------------------------------------------------
// AdaIN normalize: out_nhwc_f16 = sc*(x-mu)*rstd + bi.  Input either NCHW f32
// (in32) or NHWC f16 (in16). Coalesced NHWC writes; gather reads hit L2.
// ---------------------------------------------------------------------------
__global__ void norm_kernel(const float* __restrict__ in32,
                            const _Float16* __restrict__ in16,
                            const float* __restrict__ stats,
                            const float* __restrict__ scv,
                            const float* __restrict__ biv,
                            _Float16* __restrict__ out, int C) {
  size_t idx = ((size_t)blockIdx.x * blockDim.x + threadIdx.x) * 4;
  size_t total = (size_t)8 * HWPIX * C;
  if (idx >= total) return;
  int c0 = (int)(idx % (size_t)C);
  size_t bp = idx / (size_t)C;
  int b = (int)(bp / HWPIX);
  int p = (int)(bp % HWPIX);
#pragma unroll
  for (int k = 0; k < 4; ++k) {
    int c = c0 + k;
    float mu = stats[((size_t)b * C + c) * 2] * (1.f / (float)HWPIX);
    float vq = stats[((size_t)b * C + c) * 2 + 1] * (1.f / (float)HWPIX) - mu * mu;
    float rstd = rsqrtf(vq + 1e-5f);
    float xv = in32 ? in32[((size_t)b * C + c) * HWPIX + p] : (float)in16[idx + k];
    float yv = scv[b * C + c] * (xv - mu) * rstd + biv[b * C + c];
    out[idx + k] = (_Float16)yv;
  }
}

// ---------------------------------------------------------------------------
// Weight prep: wm[o,i,t] = wt[o,i,t]*s[b,i]*scale[i]; optional demod
// d = rsqrt(sum wm^2 + 1e-8); write f16 in WMMA A-fragment layout.
// K ordered tap-major: k = t*I + i, chunk kc = t*CI + i/32, kk = i%32.
// A-frag (16-bit 16x32): lane = (o&15) + 16*lh, lh = (kk>>3)&1,
// kk' = kk-8lh, vgpr = kk'<8 ? kk'/2 : 4+(kk'-16)/2, half = kk'&1.
// grid = (OPAD, B), block = 128
// ---------------------------------------------------------------------------
__global__ void wprep_kernel(const float* __restrict__ wt,
                             const float* __restrict__ scale,
                             const float* __restrict__ s,
                             _Float16* __restrict__ frag,
                             int O, int I, int taps, int MBL, int demod) {
  __shared__ float red[128];
  int o = blockIdx.x, b = blockIdx.y;
  int tid = threadIdx.x;
  int CI = I >> 5;
  int K = taps * I;
  int KC = taps * CI;
  size_t bstride = (size_t)KC * MBL * 512;
  _Float16* fb = frag + (size_t)b * bstride;

  auto fragpos = [&](int t, int i) -> size_t {
    int kc = t * CI + (i >> 5);
    int kk = i & 31;
    int lhh = (kk >> 3) & 1;
    int kkp = kk - (lhh << 3);
    int v = (kkp < 8) ? (kkp >> 1) : (4 + ((kkp - 16) >> 1));
    int half = kkp & 1;
    int lane = (o & 15) + (lhh << 4);
    return (((size_t)kc * MBL + (o >> 4)) * 32 + lane) * 16 + v * 2 + half;
  };

  if (o >= O) {  // zero-pad rows (rgb O=3 padded to 16)
    for (int e = tid; e < K; e += 128) { int t = e / I, i = e - t * I; fb[fragpos(t, i)] = (_Float16)0.f; }
    return;
  }
  float d = 1.f;
  if (demod) {
    float part = 0.f;
    for (int e = tid; e < K; e += 128) {
      int t = e / I, i = e - t * I;
      float wm = wt[((size_t)o * I + i) * taps + t];
      if (scale) wm *= scale[i];
      if (s)     wm *= s[b * I + i];
      part += wm * wm;
    }
    red[tid] = part; __syncthreads();
    for (int st = 64; st > 0; st >>= 1) { if (tid < st) red[tid] += red[tid + st]; __syncthreads(); }
    d = rsqrtf(red[0] + 1e-8f);
  }
  for (int e = tid; e < K; e += 128) {
    int t = e / I, i = e - t * I;
    float wm = wt[((size_t)o * I + i) * taps + t];
    if (scale) wm *= scale[i];
    if (s)     wm *= s[b * I + i];
    fb[fragpos(t, i)] = (_Float16)(wm * d);
  }
}

// ---------------------------------------------------------------------------
// Implicit-GEMM conv via v_wmma_f32_16x16x32_f16.
// Block = 256 thr (8 wave32), tile = Cout-block(16*MB) x 128 pixels (one row
// segment). Input NHWC f16; per 32-channel chunk stage 3x130 (or 1x128) halo
// tile into LDS, channel stride padded to 40 halfs (conflict-free b128).
// B-frag per lane = channels 8lh..8lh+7 and 16+8lh..23+8lh at (row,col): two ds_load_b128.
// A-frags preswizzled in global. Tap loop fully unrolled; next chunk
// prefetched (global_prefetch_b8) during compute. Epilogue: bias/leaky/
// skip-add, fused instance-norm stats (shfl + f32 atomics), f32/f16 store.
// ---------------------------------------------------------------------------
template <int TAPS, int MB>
__global__ __launch_bounds__(256) void conv_wmma_kernel(
    const _Float16* __restrict__ in, const _Float16* __restrict__ wfrag,
    size_t wf_bstride, int Cin, int Cout, int Oactive,
    const float* __restrict__ bias, const _Float16* __restrict__ addsrc16,
    float* __restrict__ stats, float* __restrict__ out32,
    _Float16* __restrict__ out16, int leaky) {
  constexpr int ROWS = (TAPS == 9) ? 3 : 1;
  constexpr int COLS = (TAPS == 9) ? 130 : 128;
  constexpr int NSLOT = ROWS * COLS;
  constexpr int CH = 40;                 // halfs per (row,col) slot (pad 32->40)
  constexpr int NGROUPS = 8 / MB;
  constexpr int NBW = 8 / NGROUPS;       // 16-pixel n-blocks per wave
  extern __shared__ _Float16 smem[];

  const int b = blockIdx.y;
  const int tile = blockIdx.x;           // 512 tiles: y = tile>>1, xhalf = tile&1
  const int y = tile >> 1;
  const int xb = (tile & 1) << 7;
  const int tid = threadIdx.x;
  const int wid = tid >> 5;
  const int lane = tid & 31;
  const int mb = wid % MB;
  const int ng = wid / MB;
  const int lh = lane >> 4;
  const int nn = lane & 15;
  const int CI = Cin >> 5;

  v8f acc[NBW] = {};
  const _Float16* inb = in + (size_t)b * HWPIX * Cin;
  const _Float16* wfb = wfrag + (size_t)b * wf_bstride;

  for (int ci = 0; ci < CI; ++ci) {
    __syncthreads();
    // stage 32 channels of the halo tile: contiguous 8B chunks from NHWC;
    // prefetch the next chunk's lines into cache while we're here.
    for (int e = tid; e < NSLOT * 8; e += 256) {
      int slot = e >> 3, part = e & 7;
      int r = slot / COLS, cc = slot - r * COLS;
      int gy = (TAPS == 9) ? (y + r - 1) : y;
      int gx = (TAPS == 9) ? (xb + cc - 1) : (xb + cc);
      unsigned long long v = 0ull;
      if (gy >= 0 && gy < 256 && gx >= 0 && gx < 256) {
        const _Float16* gp = inb + (size_t)(gy * WID + gx) * Cin + (ci << 5) + (part << 2);
        v = *(const unsigned long long*)gp;
        if (ci + 1 < CI && part == 0)
          __builtin_prefetch(gp + 32, 0, 1);  // next 32-channel chunk, same pixel
      }
      *(unsigned long long*)(smem + slot * CH + (part << 2)) = v;
    }
    __syncthreads();

#pragma unroll
    for (int t = 0; t < TAPS; ++t) {
      int kc = t * CI + ci;
      v16h a = *(const v16h*)(wfb + (((size_t)kc * MB + mb) * 32 + lane) * 16);
      int r  = (TAPS == 9) ? (t / 3) : 0;
      int dx = (TAPS == 9) ? (t % 3) : 0;
#pragma unroll
      for (int j = 0; j < NBW; ++j) {
        int cc = ng * (NBW * 16) + j * 16 + nn + dx;
        const _Float16* bp = smem + (r * COLS + cc) * CH + (lh << 3);
        v8h blo = *(const v8h*)bp;
        v8h bhi = *(const v8h*)(bp + 16);
        v16h bf;
#pragma unroll
        for (int k = 0; k < 8; ++k) { bf[k] = blo[k]; bf[k + 8] = bhi[k]; }
        acc[j] = __builtin_amdgcn_wmma_f32_16x16x32_f16(
            false, a, false, bf, (short)0, acc[j], false, false);
      }
    }
  }

  // epilogue: C/D layout -> lane nn = pixel, o = mb*16 + v + 8*lh
  float s1v[8], s2v[8];
#pragma unroll
  for (int v = 0; v < 8; ++v) { s1v[v] = 0.f; s2v[v] = 0.f; }
#pragma unroll
  for (int j = 0; j < NBW; ++j) {
    int px = ng * (NBW * 16) + j * 16 + nn;
    int p = y * WID + xb + px;
#pragma unroll
    for (int v = 0; v < 8; ++v) {
      int o = mb * 16 + v + (lh << 3);
      float val = acc[j][v];
      if (bias) val += bias[o];
      if (leaky) val = leaky_(val);
      size_t ofs = ((size_t)b * Cout + o) * HWPIX + p;
      if (addsrc16) val += (float)addsrc16[ofs];
      if (stats) { s1v[v] += val; s2v[v] += val * val; }
      if (o < Oactive) {
        if (out32) out32[ofs] = val;
        else       out16[ofs] = (_Float16)val;
      }
    }
  }
  if (stats) {
#pragma unroll
    for (int v = 0; v < 8; ++v) {
      float a1 = s1v[v], a2 = s2v[v];
      for (int m = 1; m < 16; m <<= 1) { a1 += __shfl_xor(a1, m, 32); a2 += __shfl_xor(a2, m, 32); }
      if (nn == 0) {
        int o = mb * 16 + v + (lh << 3);
        atomicAdd(&stats[((size_t)b * Cout + o) * 2],     a1);
        atomicAdd(&stats[((size_t)b * Cout + o) * 2 + 1], a2);
      }
    }
  }
}

// ---------------------------------------------------------------------------
extern "C" void kernel_launch(void* const* d_in, const int* in_sizes, int n_in,
                              void* d_out, int out_size, void* d_ws, size_t ws_size,
                              hipStream_t stream) {
  const float* x      = (const float*)d_in[0];
  const float* w      = (const float*)d_in[1];
  const float* w1     = (const float*)d_in[2];
  const float* sc1    = (const float*)d_in[3];
  const float* m1_w   = (const float*)d_in[4];
  const float* m1_b   = (const float*)d_in[5];
  const float* ln1_g  = (const float*)d_in[6];
  const float* ln1_b  = (const float*)d_in[7];
  const float* a1_sw  = (const float*)d_in[8];
  const float* a1_sb  = (const float*)d_in[9];
  const float* a1_bw  = (const float*)d_in[10];
  const float* a1_bb  = (const float*)d_in[11];
  const float* w25    = (const float*)d_in[12];
  const float* sc25   = (const float*)d_in[13];
  const float* m25_w  = (const float*)d_in[14];
  const float* m25_b  = (const float*)d_in[15];
  const float* ln25_g = (const float*)d_in[16];
  const float* ln25_b = (const float*)d_in[17];
  const float* a25_sw = (const float*)d_in[18];
  const float* a25_sb = (const float*)d_in[19];
  const float* a25_bw = (const float*)d_in[20];
  const float* a25_bb = (const float*)d_in[21];
  const float* wr     = (const float*)d_in[22];
  const float* scr    = (const float*)d_in[23];
  const float* mr_w   = (const float*)d_in[24];
  const float* mr_b   = (const float*)d_in[25];
  const float* lnr_g  = (const float*)d_in[26];
  const float* lnr_b  = (const float*)d_in[27];
  const float* ar_sw  = (const float*)d_in[28];
  const float* ar_sb  = (const float*)d_in[29];
  const float* ar_bw  = (const float*)d_in[30];
  const float* ar_bb  = (const float*)d_in[31];
  const float* sk_w   = (const float*)d_in[32];
  const float* sk_b   = (const float*)d_in[33];

  char* ws = (char*)d_ws;
  const size_t OFF_XN = 134217728;              // xup f16 NHWC 128ch at 0
  const size_t OFF_H  = 268435456;              // h f32 NCHW 64ch
  const size_t OFF_WF = 402653184;              // weight frags
  const size_t OFF_SM = OFF_WF + (4u << 20);    // stats + style scalars

  _Float16* xup    = (_Float16*)(ws);
  _Float16* xn     = (_Float16*)(ws + OFF_XN);  // normalized input (<=128ch)
  _Float16* skip16 = (_Float16*)(ws + OFF_XN + 67108864); // aliased upper half
  float*    hbuf   = (float*)(ws + OFF_H);

  _Float16* wf1 = (_Float16*)(ws + OFF_WF);     // 8 * 36*4*512 = 589824 halfs
  _Float16* wf2 = wf1 + 589824;                 // 4 layers * 294912 halfs
  _Float16* wfr = wf2 + 4 * 294912;             // 8192 halfs
  _Float16* wfs = wfr + 8192;                   // 8192 halfs

  float* smallf  = (float*)(ws + OFF_SM);
  float* stats_x = smallf;                      // 8*128*2 = 2048
  float* stats_h = smallf + 2048;               // 5 * 8*64*2 = 5120
  float* sty1    = smallf + 2048 + 5120;        // s/sc/bi 1024 each
  float* sty2    = sty1 + 3072;                 // 4 layers * (3*512)
  float* styr    = sty2 + 4 * 1536;             // 3*512

  hipMemsetAsync(smallf, 0, (2048 + 5120) * sizeof(float), stream);

  // style scalars for all 6 modulated layers
  style_kernel<<<8, 128, 0, stream>>>(w, m1_w, m1_b, ln1_g, ln1_b,
                                      a1_sw, a1_sb, a1_bw, a1_bb, 128,
                                      sty1, sty1 + 1024, sty1 + 2048);
  for (int i = 0; i < 4; ++i) {
    float* sl = sty2 + i * 1536;
    style_kernel<<<8, 64, 0, stream>>>(w, m25_w + i * 32768, m25_b + i * 64,
                                       ln25_g + i * 64, ln25_b + i * 64,
                                       a25_sw + i * 32768, a25_sb + i * 64,
                                       a25_bw + i * 32768, a25_bb + i * 64, 64,
                                       sl, sl + 512, sl + 1024);
  }
  style_kernel<<<8, 64, 0, stream>>>(w, mr_w, mr_b, lnr_g, lnr_b,
                                     ar_sw, ar_sb, ar_bw, ar_bb, 64,
                                     styr, styr + 512, styr + 1024);

  // upsample (+ stats for conv1 AdaIN)
  upsample_kernel<<<dim3(256, 8), 256, 0, stream>>>(x, xup, stats_x);

  // weight fragments (modulate + demod + swizzle)
  wprep_kernel<<<dim3(64, 8), 128, 0, stream>>>(w1, sc1, sty1, wf1, 64, 128, 9, 4, 1);
  for (int i = 0; i < 4; ++i)
    wprep_kernel<<<dim3(64, 8), 128, 0, stream>>>(w25 + i * 36864, sc25 + i * 64,
                                                  sty2 + i * 1536, wf2 + i * 294912,
                                                  64, 64, 9, 4, 1);
  wprep_kernel<<<dim3(16, 8), 128, 0, stream>>>(wr, scr, styr, wfr, 3, 64, 1, 1, 0);
  wprep_kernel<<<dim3(64, 1), 128, 0, stream>>>(sk_w, nullptr, nullptr, wfs, 64, 128, 1, 4, 0);

  // conv1: normalize xup (AdaIN) then 3x3 WMMA conv, fused leaky + stats
  norm_kernel<<<65536, 256, 0, stream>>>(nullptr, xup, stats_x,
                                         sty1 + 1024, sty1 + 2048, xn, 128);
  conv_wmma_kernel<9, 4><<<dim3(512, 8), 256, 31200, stream>>>(
      xn, wf1, (size_t)73728, 128, 64, 64, nullptr, nullptr,
      stats_h, hbuf, nullptr, 1);

  // skip: 1x1 conv on raw upsampled x (+bias), f16 NCHW output
  conv_wmma_kernel<1, 4><<<dim3(512, 8), 256, 10240, stream>>>(
      xup, wfs, (size_t)0, 128, 64, 64, sk_b, nullptr,
      nullptr, nullptr, skip16, 0);

  // conv2..conv5; conv5 adds skip and writes h directly into d_out
  for (int i = 0; i < 4; ++i) {
    float* sl = sty2 + i * 1536;
    norm_kernel<<<32768, 256, 0, stream>>>(hbuf, nullptr,
                                           stats_h + i * 1024,
                                           sl + 512, sl + 1024, xn, 64);
    float* hout = (i == 3) ? (float*)d_out : hbuf;
    conv_wmma_kernel<9, 4><<<dim3(512, 8), 256, 31200, stream>>>(
        xn, wf2 + i * 294912, (size_t)36864, 64, 64, 64, nullptr,
        (i == 3) ? skip16 : nullptr, stats_h + (i + 1) * 1024, hout, nullptr, 1);
  }

  // rgb: AdaIN on h, 1x1 modconv (no demod/leaky), Cout=3 (padded M=16)
  norm_kernel<<<32768, 256, 0, stream>>>((const float*)d_out, nullptr,
                                         stats_h + 4 * 1024,
                                         styr + 512, styr + 1024, xn, 64);
  conv_wmma_kernel<1, 1><<<dim3(512, 8), 256, 10240, stream>>>(
      xn, wfr, (size_t)1024, 64, 3, 3, nullptr, nullptr, nullptr,
      (float*)d_out + 33554432, nullptr, 0);
}
